// ConvCaps_8976481649049
// MI455X (gfx1250) — matrix operand order repaired and checked by reading, not compiled, not run on hardware
//
#include <hip/hip_runtime.h>

typedef __attribute__((ext_vector_type(16))) __bf16 v16bf;
typedef __attribute__((ext_vector_type(8)))  float  v8f;

// ---------------- geometry ----------------
// x:      [8][8][16][64][64] f32
// conv_w: [256][16][3][3]    f32
// out:    [8][16][16][64][64] f32
// GEMM per (b,ci): A[16 pix][K=144->160] * B[K][256] ; 5 k-steps of 16x16x32 bf16 WMMA
#define KSTEPS 5

// ---- dynamic LDS partition (bytes) ----
#define VOTES_OFF 0         // f32 [16 pix][8 ci][16 co][20 pad]  = 163840 B
#define AFRAG_OFF 163840    // u16 [8 ci][5 ks][32 lane][16 e]    = 40960 B
#define XS_OFF    204800    // u16 [8 ci][16 pi][3 ky][20 pad]    = 15360 B
#define VJS_OFF   220160    // f32 [256 cp][18 pad]               = 18432 B
#define SMEM_BYTES 238592

static __device__ __forceinline__ unsigned short f2bf(float f){
  unsigned int u = __float_as_uint(f);
  u += 0x7fffu + ((u >> 16) & 1u);          // round to nearest even
  return (unsigned short)(u >> 16);
}

// ============ pre-pass: conv_w -> bf16 WMMA B-fragments in d_ws ============
// Layout: bFragG[ks][nt][lane][e], 5*16*32*16 halves (80 KB).
// B[k][n]: k = (ky*3+kx)*16 + pi ; n = out-channel (co*16+po).
// B lane map used here (contiguous k, same as A): n = lane&15, K = ks*32 + (lane>>4)*16 + e.
__global__ void bfrag_prep_kernel(const float* __restrict__ w,
                                  unsigned short* __restrict__ bFragG){
  int i = blockIdx.x * 256 + threadIdx.x;
  if (i >= KSTEPS*16*32*16) return;
  int e    = i & 15;
  int lane = (i >> 4) & 31;
  int nt   = (i >> 9) & 15;
  int ks   = i >> 13;
  int k = ks*32 + ((lane >> 4) << 4) + e;
  int n = nt*16 + (lane & 15);
  float v = 0.0f;
  if (k < 144){
    int sp = k >> 4;           // ky*3+kx
    int pi = k & 15;
    int ky = sp / 3;
    int kx = sp - ky*3;
    v = w[((n*16 + pi)*3 + ky)*3 + kx];
  }
  bFragG[i] = f2bf(v);
}

// ============ fused conv(WMMA) + dynamic routing ============
__global__ __launch_bounds__(256)
void convcaps_fused_kernel(const float* __restrict__ x,
                           const unsigned short* __restrict__ bFragG,
                           const float* __restrict__ biases,
                           const int* __restrict__ routings_p,
                           float* __restrict__ out){
  extern __shared__ char smem[];
  float*          votes = (float*)(smem + VOTES_OFF);          // [16][8][16][20]
  unsigned short* aFrag = (unsigned short*)(smem + AFRAG_OFF); // [8][5][32][16]
  unsigned short* xs    = (unsigned short*)(smem + XS_OFF);    // [8][16][3][20]
  float*          vjS   = (float*)(smem + VJS_OFF);            // [256][18]

  const int tid  = threadIdx.x;
  const int wg   = blockIdx.x;        // 2048 = 8 b * 64 y * 4 xtile
  const int b    = wg >> 8;
  const int y    = (wg >> 2) & 63;
  const int x0   = (wg & 3) << 4;

  // ---- Phase 1: stage x halo rows (bf16) : xs[ci][pi][ky][xi 0..17] ----
  const float* xb = x + (size_t)b * (8*16*4096);
  for (int i = tid; i < 8*16*3*18; i += 256){
    int row = i / 18;                 // (ci*16+pi)*3+ky
    int xi  = i - row*18;
    int ci  = row / 48;
    int r2  = row - ci*48;
    int pi  = r2 / 3;
    int ky  = r2 - pi*3;
    int yy  = y + ky - 1;
    int xx  = x0 + xi - 1;
    float v = 0.0f;
    if (yy >= 0 && yy < 64 && xx >= 0 && xx < 64)
      v = xb[(ci*16 + pi)*4096 + yy*64 + xx];
    xs[row*20 + xi] = f2bf(v);
  }
  __syncthreads();

  // ---- Phase 2: build A-fragments in WMMA lane order ----
  // A[m][k], m = pixel; lane map: m = lane&15, K = ks*32 + (lane>>4)*16 + e
  for (int i = tid; i < 8*KSTEPS*32*16; i += 256){
    int e    = i & 15;
    int lane = (i >> 4) & 31;
    int r    = i >> 9;                // ci*5 + ks
    int ks   = r % 5;
    int ci   = r / 5;
    int k = ks*32 + ((lane >> 4) << 4) + e;
    unsigned short v = 0;
    if (k < 144){
      int sp = k >> 4;
      int pi = k & 15;
      int ky = sp / 3;
      int kx = sp - ky*3;
      int m  = lane & 15;
      v = xs[((ci*16 + pi)*3 + ky)*20 + m + kx];
    }
    aFrag[i] = v;
  }
  __syncthreads();

  // ---- Phase 3: WMMA — votes[pix][ci][co][po] ----
  const int wv   = tid >> 5;          // wave 0..7
  const int lane = tid & 31;
  const int half = lane >> 4;
  const int ln   = lane & 15;

  for (int nt2 = 0; nt2 < 2; ++nt2){
    const int nt = wv*2 + nt2;        // n-tile == co
    v16bf bfr[KSTEPS];
#pragma unroll
    for (int ks = 0; ks < KSTEPS; ++ks)
      bfr[ks] = *(const v16bf*)(bFragG + (((ks*16 + nt)*32 + lane) << 4));
    for (int ci = 0; ci < 8; ++ci){
      v8f acc = {0.f,0.f,0.f,0.f,0.f,0.f,0.f,0.f};
#pragma unroll
      for (int ks = 0; ks < KSTEPS; ++ks){
        v16bf a = *(const v16bf*)(aFrag + (((ci*5 + ks)*32 + lane) << 4));
        acc = __builtin_amdgcn_wmma_f32_16x16x32_bf16(
                  false, a, false, bfr[ks], (short)0, acc, false, false);
      }
#pragma unroll
      for (int r = 0; r < 8; ++r){
        int m = r + half*8;           // pixel, D-layout: M = r + (lane>>4)*8
        votes[((m*8 + ci)*16 + nt)*20 + ln] = acc[r];
      }
    }
  }
  __syncthreads();

  // ---- Phase 4: dynamic routing; thread = (pixel p, out-capsule co) ----
  const int p  = wv*2 + half;
  const int co = ln;
  const int R  = routings_p[0];

  float biasr[16];
#pragma unroll
  for (int q = 0; q < 4; ++q){
    float4 b4 = ((const float4*)(biases + co*16))[q];
    biasr[q*4+0] = b4.x; biasr[q*4+1] = b4.y; biasr[q*4+2] = b4.z; biasr[q*4+3] = b4.w;
  }

  float bij[8] = {0,0,0,0,0,0,0,0};
  float vj[16];
#pragma unroll
  for (int po = 0; po < 16; ++po) vj[po] = 0.f;

  for (int it = 0; it < R; ++it){
    float sj[16];
#pragma unroll
    for (int po = 0; po < 16; ++po) sj[po] = biasr[po];
    for (int ci = 0; ci < 8; ++ci){
      float bv = bij[ci];
      float mx = bv;                          // softmax over the 16 co lanes
      mx = fmaxf(mx, __shfl_xor(mx, 1, 16));
      mx = fmaxf(mx, __shfl_xor(mx, 2, 16));
      mx = fmaxf(mx, __shfl_xor(mx, 4, 16));
      mx = fmaxf(mx, __shfl_xor(mx, 8, 16));
      float e = __expf(bv - mx);
      float s = e;
      s += __shfl_xor(s, 1, 16);
      s += __shfl_xor(s, 2, 16);
      s += __shfl_xor(s, 4, 16);
      s += __shfl_xor(s, 8, 16);
      float cij = e / s;
      const float* vr = votes + ((p*8 + ci)*16 + co)*20;
#pragma unroll
      for (int q = 0; q < 4; ++q){
        float4 v4 = *(const float4*)(vr + q*4);
        sj[q*4+0] += cij * v4.x; sj[q*4+1] += cij * v4.y;
        sj[q*4+2] += cij * v4.z; sj[q*4+3] += cij * v4.w;
      }
    }
    float n2 = 0.f;
#pragma unroll
    for (int po = 0; po < 16; ++po) n2 += sj[po]*sj[po];
    float scale = sqrtf(n2) / (1.0f + n2);    // squash: |s|/(1+|s|^2)
#pragma unroll
    for (int po = 0; po < 16; ++po) vj[po] = scale * sj[po];

    if (it < R - 1){
      for (int ci = 0; ci < 8; ++ci){
        const float* vr = votes + ((p*8 + ci)*16 + co)*20;
        float dot = 0.f;
#pragma unroll
        for (int q = 0; q < 4; ++q){
          float4 v4 = *(const float4*)(vr + q*4);
          dot += v4.x*vj[q*4+0] + v4.y*vj[q*4+1] + v4.z*vj[q*4+2] + v4.w*vj[q*4+3];
        }
        bij[ci] += dot;
      }
    }
  }

  // ---- Phase 5: stage vj, then coalesced global store ----
#pragma unroll
  for (int po = 0; po < 16; ++po)
    vjS[(co*16 + po)*18 + p] = vj[po];
  __syncthreads();

  float* ob = out + (size_t)b*256*4096 + y*64 + x0;
  for (int c = 0; c < 16; ++c){
    int cp   = c*16 + (tid >> 4);     // co*16+po
    int xoff = tid & 15;
    ob[(size_t)cp*4096 + xoff] = vjS[cp*18 + xoff];
  }
}

extern "C" void kernel_launch(void* const* d_in, const int* in_sizes, int n_in,
                              void* d_out, int out_size, void* d_ws, size_t ws_size,
                              hipStream_t stream) {
  (void)in_sizes; (void)n_in; (void)out_size; (void)ws_size;
  const float* x        = (const float*)d_in[0];
  const float* conv_w   = (const float*)d_in[1];
  const float* biases   = (const float*)d_in[2];
  const int*   routings = (const int*)d_in[3];
  float*       out      = (float*)d_out;
  unsigned short* bFragG = (unsigned short*)d_ws;   // 80 KB

  (void)hipFuncSetAttribute((const void*)convcaps_fused_kernel,
                            hipFuncAttributeMaxDynamicSharedMemorySize,
                            SMEM_BYTES);

  bfrag_prep_kernel<<<(KSTEPS*16*32*16 + 255)/256, 256, 0, stream>>>(conv_w, bFragG);
  convcaps_fused_kernel<<<2048, 256, SMEM_BYTES, stream>>>(x, bFragG, biases, routings, out);
}
